// one_batch_net_69071664054902
// MI455X (gfx1250) — matrix operand order repaired and checked by reading, not compiled, not run on hardware
//
#include <hip/hip_runtime.h>

typedef __attribute__((ext_vector_type(16))) _Float16 v16h;
typedef __attribute__((ext_vector_type(8)))  _Float16 v8h;
typedef __attribute__((ext_vector_type(8)))  float    v8f;

#define NB    2
#define NN    512
#define NSEL  128
#define NTYPE 4
#define F0    32
#define F1    64
#define F2    128
#define DIN   2048   // 16 * 128
#define HH    256

// branch-free tanh on the native transcendental pipe:
// tanh(x) = 1 - 2 / (exp2(2*log2(e)*x) + 1); saturates to +-1, NaN propagates.
__device__ __forceinline__ float fast_tanh(float x) {
  float e2 = __builtin_amdgcn_exp2f(x * 2.8853900817779268f);
  return 1.0f - 2.0f * __builtin_amdgcn_rcpf(e2 + 1.0f);
}

__device__ __forceinline__ v16h cat8(v8h lo, v8h hi) {
  return __builtin_shufflevector(lo, hi, 0, 1, 2, 3, 4, 5, 6, 7,
                                         8, 9, 10, 11, 12, 13, 14, 15);
}

// A fragment (16x32 f16) from row-major LDS tile [row][col], ld in halves.
// Wave32 layout: lane m=lane&15 row; elems 0..7 = K hi*8.., elems 8..15 = K 16+hi*8..
// -> two contiguous 16B loads (ds_load_b128 x2).
__device__ __forceinline__ v16h load_a_lds(const _Float16* S, int row0, int col0, int ld) {
  int lane = threadIdx.x & 31;
  int m = lane & 15, hi = lane >> 4;
  const _Float16* rp = S + (size_t)(row0 + m) * ld + col0 + hi * 8;
  v8h lo  = *(const v8h*)rp;
  v8h hi8 = *(const v8h*)(rp + 16);
  return cat8(lo, hi8);
}

// B fragment from pre-swizzled global weights: 512 halves per 32x16 fragment,
// per-lane 16 operand halves contiguous -> two global_load_b128.
__device__ __forceinline__ v16h load_b_frag(const _Float16* base) {
  int lane = threadIdx.x & 31;
  const _Float16* p = base + lane * 16;
  v8h lo  = *(const v8h*)p;
  v8h hi8 = *(const v8h*)(p + 8);
  return cat8(lo, hi8);
}

__device__ __forceinline__ v8f wmma_f16(v16h a, v16h b, v8f c) {
  return __builtin_amdgcn_wmma_f32_16x16x32_f16(false, a, false, b, (short)0, c, false, false);
}

// offset (in halves) of element (k,n) inside a K x N matrix stored as B fragments
__device__ __forceinline__ size_t bfrag_off(int k, int n, int Ntot) {
  int kt = k >> 5, nt = n >> 4;
  int lane = (n & 15) + (((k >> 3) & 1) << 4);
  int e = (k & 7) + (((k >> 4) & 1) << 3);
  return ((((size_t)kt * (Ntot >> 4) + nt) << 9)) + (size_t)lane * 16 + e;
}

// ---------- kernel: convert f32 weights to f16 B-fragment layout ----------
__global__ void prep_bfrag_kernel(const float* __restrict__ src, _Float16* __restrict__ dst,
                                  int nmat, int K, int N) {
  int total = nmat * K * N;
  for (int i = blockIdx.x * blockDim.x + threadIdx.x; i < total;
       i += gridDim.x * blockDim.x) {
    int m = i / (K * N);
    int rem = i - m * (K * N);
    int k = rem / N;
    int n = rem - k * N;
    dst[(size_t)m * K * N + bfrag_off(k, n, N)] = (_Float16)src[i];
  }
}

// ---------- kernel: zero output accumulators ----------
__global__ void zero_out_kernel(float* out) {
  if (threadIdx.x < NB) out[threadIdx.x] = 0.0f;
}

// ---------- kernel 1: per-atom embedding net + descriptor D ----------
__global__ __launch_bounds__(256) void embed_kernel(
    const float* __restrict__ sym,  const float* __restrict__ stdv,
    const float* __restrict__ avgv, const int* __restrict__ atype,
    const int* __restrict__ ntype,
    const float* __restrict__ Wf0, const float* __restrict__ bf0,
    const _Float16* __restrict__ Wf1f, const float* __restrict__ bf1,
    const _Float16* __restrict__ Wf2f, const float* __restrict__ bf2,
    _Float16* __restrict__ Dws)
{
  __shared__ float    x_s[NSEL][4];
  __shared__ int      nt_s[NSEL];
  __shared__ alignas(16) _Float16 h0_s[NSEL][F0];
  __shared__ alignas(16) _Float16 h1_s[NSEL][F1];
  __shared__ alignas(16) _Float16 G_s[NSEL][F2];
  __shared__ float    GR_s[F2][4];

  const int a   = blockIdx.x;             // atom index (b*NN + n)
  const int tid = threadIdx.x;
  const int at  = atype[a];

  float avg4[4], std4[4];
#pragma unroll
  for (int d = 0; d < 4; ++d) { avg4[d] = avgv[at * 4 + d]; std4[d] = stdv[at * 4 + d]; }

  for (int i = tid; i < NSEL * 4; i += 256) {
    int s = i >> 2, d = i & 3;
    float v = sym[((size_t)a * NSEL + s) * 4 + d];
    x_s[s][d] = (v - avg4[d]) / std4[d];
  }
  for (int i = tid; i < NSEL; i += 256) nt_s[i] = ntype[(size_t)a * NSEL + i];
  __syncthreads();

  const int w    = tid >> 5;
  const int lane = tid & 31;
  const int col  = lane & 15;
  const int hi   = lane >> 4;
  const int row0 = w * 16;                // each of the 8 waves owns 16 neighbors

  for (int t = 0; t < NTYPE; ++t) {
    const int p = at * NTYPE + t;
    // h0 = tanh(s * Wf0[p] + bf0[p])
    for (int i = tid; i < NSEL * F0; i += 256) {
      int s = i >> 5, c = i & 31;
      h0_s[s][c] = (_Float16)fast_tanh(x_s[s][0] * Wf0[p * F0 + c] + bf0[p * F0 + c]);
    }
    __syncthreads();

    // layer 1: (128x32) @ (32x64) -> h1
    {
      v16h af = load_a_lds(&h0_s[0][0], row0, 0, F0);
      v8f acc[4];
#pragma unroll
      for (int j = 0; j < 4; ++j) {
        v16h bf = load_b_frag(Wf1f + (((size_t)p * 4 + j) << 9));
        v8f z = {};
        acc[j] = wmma_f16(af, bf, z);
      }
#pragma unroll
      for (int j = 0; j < 4; ++j) {
        int n0 = j * 16;
        float bb = bf1[p * F1 + n0 + col];
#pragma unroll
        for (int r = 0; r < 8; ++r) {
          int m = row0 + r + hi * 8;
          h1_s[m][n0 + col] = (_Float16)fast_tanh(acc[j][r] + bb);
        }
      }
    }
    __syncthreads();

    // layer 2: (128x64) @ (64x128) -> masked tanh write into G
    {
      v16h af0 = load_a_lds(&h1_s[0][0], row0, 0, F1);
      v16h af1 = load_a_lds(&h1_s[0][0], row0, 32, F1);
      v8f acc[8];
#pragma unroll
      for (int j = 0; j < 8; ++j) {
        v8f z = {};
        v16h b0 = load_b_frag(Wf2f + (((size_t)p * 16 + j) << 9));
        z = wmma_f16(af0, b0, z);
        v16h b1 = load_b_frag(Wf2f + (((size_t)p * 16 + 8 + j) << 9));
        acc[j] = wmma_f16(af1, b1, z);
      }
#pragma unroll
      for (int j = 0; j < 8; ++j) {
        int n0 = j * 16;
        float bb = bf2[p * F2 + n0 + col];
#pragma unroll
        for (int r = 0; r < 8; ++r) {
          int m = row0 + r + hi * 8;
          if (nt_s[m] == t)   // each neighbor row matches exactly one t
            G_s[m][n0 + col] = (_Float16)fast_tanh(acc[j][r] + bb);
        }
      }
    }
    __syncthreads();
  }

  // GR[m][d] = (1/S) * sum_s G[s][m] * x[s][d]
  for (int i = tid; i < F2 * 4; i += 256) {
    int m = i >> 2, d = i & 3;
    float sum = 0.0f;
    for (int s = 0; s < NSEL; ++s) sum += (float)G_s[s][m] * x_s[s][d];
    GR_s[m][d] = sum * (1.0f / NSEL);
  }
  __syncthreads();

  // D[m][k] = dot4(GR[m], GR[k]) for k<16 ; store f16 descriptor to workspace
  for (int i = tid; i < DIN; i += 256) {
    int m = i >> 4, k = i & 15;
    float sum = 0.0f;
#pragma unroll
    for (int d = 0; d < 4; ++d) sum += GR_s[m][d] * GR_s[k][d];
    Dws[(size_t)a * DIN + i] = (_Float16)sum;
  }
}

// ---------- kernel 2: fitting net over 16-atom tiles ----------
__global__ __launch_bounds__(128) void fit_kernel(
    const _Float16* __restrict__ Dws, const int* __restrict__ atype,
    const _Float16* __restrict__ Wg0f, const float* __restrict__ bg0,
    const _Float16* __restrict__ Wg1f, const float* __restrict__ bg1,
    const _Float16* __restrict__ Wg2f, const float* __restrict__ bg2,
    const float* __restrict__ Wgo, const float* __restrict__ bgo,
    float* __restrict__ out)
{
  __shared__ alignas(16) _Float16 A0[16][DIN];    // 64 KB: 16 atoms x 2048
  __shared__ alignas(16) _Float16 Hh[2][16][HH];  // 16 KB ping-pong hidden
  __shared__ int at_s[16];

  const int tid = threadIdx.x;
  const int a0  = blockIdx.x * 16;

  for (int i = tid; i < 16 * DIN; i += 128)
    A0[i >> 11][i & (DIN - 1)] = Dws[(size_t)(a0 + (i >> 11)) * DIN + (i & (DIN - 1))];
  if (tid < 16) at_s[tid] = atype[a0 + tid];
  __syncthreads();

  const int w    = tid >> 5;
  const int lane = tid & 31;
  const int col  = lane & 15;
  const int hi   = lane >> 4;

  for (int t = 0; t < NTYPE; ++t) {
    // layer g0: (16 x 2048) @ (2048 x 256); each wave owns 4 N-tiles
    {
      v8f acc[4] = {};
      for (int ks = 0; ks < DIN / 32; ++ks) {
        v16h af = load_a_lds(&A0[0][0], 0, ks * 32, DIN);
#pragma unroll
        for (int j = 0; j < 4; ++j) {
          int ntile = w + 4 * j;
          v16h bf = load_b_frag(Wg0f + (((size_t)(t * 64 + ks) * 16 + ntile) << 9));
          acc[j] = wmma_f16(af, bf, acc[j]);
        }
      }
#pragma unroll
      for (int j = 0; j < 4; ++j) {
        int n0 = (w + 4 * j) * 16;
        float bb = bg0[t * HH + n0 + col];
#pragma unroll
        for (int r = 0; r < 8; ++r) {
          int m = r + hi * 8;
          Hh[0][m][n0 + col] = (_Float16)fast_tanh(acc[j][r] + bb);
        }
      }
    }
    __syncthreads();

    // layers g1, g2: (16 x 256) @ (256 x 256), ping-pong
    for (int layer = 0; layer < 2; ++layer) {
      const _Float16* Wlf = (layer == 0) ? Wg1f : Wg2f;
      const float*    bl  = (layer == 0) ? bg1 : bg2;
      const _Float16* src = &Hh[layer][0][0];
      _Float16*       dst = &Hh[1 - layer][0][0];
      v8f acc[4] = {};
#pragma unroll
      for (int ks = 0; ks < HH / 32; ++ks) {
        v16h af = load_a_lds(src, 0, ks * 32, HH);
#pragma unroll
        for (int j = 0; j < 4; ++j) {
          int ntile = w + 4 * j;
          v16h bf = load_b_frag(Wlf + (((size_t)(t * 8 + ks) * 16 + ntile) << 9));
          acc[j] = wmma_f16(af, bf, acc[j]);
        }
      }
#pragma unroll
      for (int j = 0; j < 4; ++j) {
        int n0 = (w + 4 * j) * 16;
        float bb = bl[t * HH + n0 + col];
#pragma unroll
        for (int r = 0; r < 8; ++r) {
          int m = r + hi * 8;
          dst[m * HH + n0 + col] = (_Float16)fast_tanh(acc[j][r] + bb);
        }
      }
      __syncthreads();
    }

    // output layer: e = dot(H, Wgo[t]) + bgo[t]; masked accumulate per batch
    {
      int am  = tid >> 3;   // 16 atoms x 8 threads
      int sub = tid & 7;
      float partial = 0.0f;
      for (int k = sub * 32; k < sub * 32 + 32; ++k)
        partial += (float)Hh[0][am][k] * Wgo[t * HH + k];
      partial += __shfl_down(partial, 4, 8);
      partial += __shfl_down(partial, 2, 8);
      partial += __shfl_down(partial, 1, 8);
      if (sub == 0 && at_s[am] == t) {
        int aa = a0 + am;
        atomicAdd(&out[aa / NN], partial + bgo[t]);
      }
    }
    __syncthreads();
  }
}

extern "C" void kernel_launch(void* const* d_in, const int* in_sizes, int n_in,
                              void* d_out, int out_size, void* d_ws, size_t ws_size,
                              hipStream_t stream) {
  const float* sym   = (const float*)d_in[0];
  const float* stdv  = (const float*)d_in[1];
  const float* avgv  = (const float*)d_in[2];
  const int*   atype = (const int*)d_in[3];
  const int*   ntype = (const int*)d_in[4];
  const float* Wf0   = (const float*)d_in[5];
  const float* bf0   = (const float*)d_in[6];
  const float* Wf1   = (const float*)d_in[7];
  const float* bf1   = (const float*)d_in[8];
  const float* Wf2   = (const float*)d_in[9];
  const float* bf2   = (const float*)d_in[10];
  const float* Wg0   = (const float*)d_in[11];
  const float* bg0   = (const float*)d_in[12];
  const float* Wg1   = (const float*)d_in[13];
  const float* bg1   = (const float*)d_in[14];
  const float* Wg2   = (const float*)d_in[15];
  const float* bg2   = (const float*)d_in[16];
  const float* Wgo   = (const float*)d_in[17];
  const float* bgo   = (const float*)d_in[18];
  float* out = (float*)d_out;

  // workspace layout (halves)
  _Float16* base = (_Float16*)d_ws;
  size_t off = 0;
  _Float16* Dws  = base + off; off += (size_t)NB * NN * DIN;        // 4 MB
  _Float16* Wf1f = base + off; off += (size_t)16 * F0 * F1;         // 64 KB
  _Float16* Wf2f = base + off; off += (size_t)16 * F1 * F2;         // 256 KB
  _Float16* Wg0f = base + off; off += (size_t)NTYPE * DIN * HH;     // 4 MB
  _Float16* Wg1f = base + off; off += (size_t)NTYPE * HH * HH;      // 512 KB
  _Float16* Wg2f = base + off; off += (size_t)NTYPE * HH * HH;      // 512 KB

  // pre-swizzle all weight matrices into WMMA B-fragment layout (f16)
  prep_bfrag_kernel<<<(16 * F0 * F1 + 255) / 256, 256, 0, stream>>>(Wf1, Wf1f, 16, F0, F1);
  prep_bfrag_kernel<<<(16 * F1 * F2 + 255) / 256, 256, 0, stream>>>(Wf2, Wf2f, 16, F1, F2);
  prep_bfrag_kernel<<<(NTYPE * DIN * HH + 255) / 256, 256, 0, stream>>>(Wg0, Wg0f, NTYPE, DIN, HH);
  prep_bfrag_kernel<<<(NTYPE * HH * HH + 255) / 256, 256, 0, stream>>>(Wg1, Wg1f, NTYPE, HH, HH);
  prep_bfrag_kernel<<<(NTYPE * HH * HH + 255) / 256, 256, 0, stream>>>(Wg2, Wg2f, NTYPE, HH, HH);

  zero_out_kernel<<<1, 32, 0, stream>>>(out);
  embed_kernel<<<NB * NN, 256, 0, stream>>>(sym, stdv, avgv, atype, ntype,
                                            Wf0, bf0, Wf1f, bf1, Wf2f, bf2, Dws);
  fit_kernel<<<(NB * NN) / 16, 128, 0, stream>>>(Dws, atype, Wg0f, bg0, Wg1f, bg1,
                                                 Wg2f, bg2, Wgo, bgo, out);
}
